// RSSM_43903155700503
// MI455X (gfx1250) — matrix operand order repaired and checked by reading, not compiled, not run on hardware
//
#include <hip/hip_runtime.h>
#include <hip/hip_bf16.h>

// ---------------------------------------------------------------------------
// RSSM scan for MI455X (gfx1250): 32 workgroups x 512 threads (16 waves).
// Each workgroup owns a 16-row batch tile and runs all 64 timesteps.
// GRU + posterior matmuls use v_wmma_f32_16x16x32_bf16; the one-hot input
// matmul is a column gather from a pre-transposed W_in table.
// Weight fragments are re-streamed from the L2-resident bf16 tables every
// timestep; an asm optimization barrier stops LICM from hoisting (and then
// spilling) the loop-invariant weight loads.
// ---------------------------------------------------------------------------

typedef __bf16 bf16_t;
typedef __attribute__((ext_vector_type(16))) __bf16 v16bf;
typedef __attribute__((ext_vector_type(8)))  float  v8f;
typedef __attribute__((ext_vector_type(4)))  float  f32x4;

#define B_TOT   512
#define T_STEPS 64
#define H_DIM   256
#define A_DIM   6
#define R_DIM   256
#define L_DIM   1024
#define MB      16     // batch rows per workgroup (WMMA M)
#define NTHREADS 512   // 16 waves

// padded LDS row strides (elements) to rotate banks across rows
#define SX 264         // 256 + 8   (bf16 / f32 rows of width 256)
#define SH 528         // 512 + 16  (concat [h|obs] bf16 rows)
#define SL 1040        // 1024 + 16 (logits+gumbel f32 rows)

// ---- WMMA fragment loaders (contiguous k order, same in A and B) ------------------

// A-fragment 16x32 bf16 from LDS row-major (stride in elements).
// lane 0-15: M=lane, K = kbase+{0..15}; lane 16-31: same M, K = kbase+{16..31}.
__device__ __forceinline__ v16bf load_a_frag(const bf16_t* base, int stride,
                                             int kbase, int lane) {
    int m  = lane & 15;
    int kb = (lane >> 4) << 4;               // 0 or 16
    const bf16_t* p = base + m * stride + kbase + kb;
    v16bf r;
    f32x4* r4 = reinterpret_cast<f32x4*>(&r);
    r4[0] = *reinterpret_cast<const f32x4*>(p);        // K = kbase+kb .. +7
    r4[1] = *reinterpret_cast<const f32x4*>(p + 8);    // K = kbase+8+kb .. +7
    return r;
}

// B-fragment 32x16 bf16 from global weight matrix stored row-major as (N x K):
// out[m][n] = sum_k A[m][k] * W[n][k].  lane = N, lane-half selects K 0-15/16-31,
// which is 16 *contiguous* bf16 per lane = two b128 loads.
__device__ __forceinline__ v16bf load_b_frag(const bf16_t* __restrict__ W, int kdim,
                                             int n0, int kbase, int lane) {
    int n  = n0 + (lane & 15);
    int kh = (lane >> 4) << 4;               // 0 or 16
    const bf16_t* p = W + (size_t)n * kdim + kbase + kh;
    v16bf r;
    f32x4* r4 = reinterpret_cast<f32x4*>(&r);
    r4[0] = *reinterpret_cast<const f32x4*>(p);
    r4[1] = *reinterpret_cast<const f32x4*>(p + 8);
    return r;
}

__device__ __forceinline__ v8f wmma_bf16(v16bf a, v16bf b, v8f c) {
    return __builtin_amdgcn_wmma_f32_16x16x32_bf16(
        /*neg_a=*/false, a, /*neg_b=*/false, b,
        /*c_mod=*/(short)0, c, /*reuse_a=*/false, /*reuse_b=*/false);
}

__device__ __forceinline__ float sigmoidf_(float x) {
    return 1.0f / (1.0f + __expf(-x));
}

// ---- preamble: weights f32 -> bf16 (row-major), W_in -> transposed f32 -----
__global__ void rssm_prep_weights(const float* __restrict__ W_ih,
                                  const float* __restrict__ W_hh,
                                  const float* __restrict__ W_post,
                                  const float* __restrict__ W_in,
                                  bf16_t* __restrict__ wih,
                                  bf16_t* __restrict__ whh,
                                  bf16_t* __restrict__ wpost,
                                  float*  __restrict__ winT) {
    int idx = blockIdx.x * blockDim.x + threadIdx.x;
    int stride = gridDim.x * blockDim.x;
    for (int i = idx; i < 768 * 256; i += stride) {
        wih[i] = (bf16_t)W_ih[i];
        whh[i] = (bf16_t)W_hh[i];
    }
    for (int i = idx; i < 1024 * 512; i += stride)
        wpost[i] = (bf16_t)W_post[i];
    // W_in is (H=256, L+A=1030) row-major; store columns contiguously:
    // winT[c*256 + h] = W_in[h*1030 + c]  -> one-hot gather becomes coalesced.
    for (int i = idx; i < 1030 * 256; i += stride) {
        int c = i >> 8, h = i & 255;
        winT[i] = W_in[h * 1030 + c];
    }
}

// ---- main persistent scan kernel -------------------------------------------
__global__ void __launch_bounds__(NTHREADS, 1)
rssm_scan_kernel(const float* __restrict__ obs,
                 const float* __restrict__ action,
                 const unsigned char* __restrict__ first,
                 const float* __restrict__ gumbel,
                 const float* __restrict__ b_in,
                 const float* __restrict__ b_ih,
                 const float* __restrict__ b_hh,
                 const float* __restrict__ b_post,
                 const bf16_t* __restrict__ wih,
                 const bf16_t* __restrict__ whh,
                 const bf16_t* __restrict__ wpost,
                 const float* __restrict__ winT,
                 float* __restrict__ deter_out,
                 float* __restrict__ stoch_out,
                 float* __restrict__ logits_out) {
    __shared__ float  deter[MB][SX];     // GRU hidden state (f32, persistent)
    __shared__ bf16_t xbf[MB][SX];       // in-linear output, bf16 A-tile
    __shared__ bf16_t dbf[MB][SX];       // deter as bf16 A-tile
    __shared__ bf16_t habf[MB][SH];      // concat [h | obs] bf16 A-tile
    __shared__ float  lg[MB][SL];        // logits + gumbel
    __shared__ int    sidx[MB][32];      // one-hot index per group (-1 == zeros)
    __shared__ float  actl[MB][A_DIM];
    __shared__ int    fflag[MB];

    const int tid   = threadIdx.x;
    const int lane  = tid & 31;
    const int wave  = tid >> 5;          // 0..15
    const int b0    = blockIdx.x * MB;
    const int mbase = (lane >> 4) << 3;  // C-fragment row base: 0 or 8

    // carry0 = zeros: deter = 0, stoch = all-zero (idx = -1)
    for (int e = tid; e < MB * H_DIM; e += NTHREADS)
        deter[e >> 8][e & 255] = 0.0f;
    if (tid < MB * 32) sidx[tid >> 5][tid & 31] = -1;
    __syncthreads();

#pragma unroll 1
    for (int t = 0; t < T_STEPS; ++t) {
        // Optimization barrier: make the weight base pointers opaque per
        // timestep so LICM cannot hoist (and the RA then spill) the
        // loop-invariant B-fragment loads out of the time loop.
        int tz = 0;
        asm volatile("" : "+s"(tz));
        const bf16_t* wih_t   = wih   + tz;
        const bf16_t* whh_t   = whh   + tz;
        const bf16_t* wpost_t = wpost + tz;

        // ---- stage per-step inputs; apply episode reset -------------------
        if (tid < MB) fflag[tid] = first[(b0 + tid) * T_STEPS + t] ? 1 : 0;
        if (tid < MB * A_DIM) {
            int m = tid / A_DIM, j = tid % A_DIM;
            actl[m][j] = action[((size_t)(b0 + m) * T_STEPS + t) * A_DIM + j];
        }
        __syncthreads();
        if (tid < MB * 32 && fflag[tid >> 5]) sidx[tid >> 5][tid & 31] = -1;
        for (int e = tid; e < MB * H_DIM; e += NTHREADS) {
            int m = e >> 8, h = e & 255;
            float d = fflag[m] ? 0.0f : deter[m][h];
            deter[m][h] = d;
            dbf[m][h]   = (bf16_t)d;
            habf[m][256 + h] =
                (bf16_t)obs[((size_t)(b0 + m) * T_STEPS + t) * H_DIM + h];
        }
        __syncthreads();

        // ---- x = b_in + one-hot column gather + action low-rank term ------
        for (int e = tid; e < MB * H_DIM; e += NTHREADS) {
            int m = e >> 8, h = e & 255;
            float v = b_in[h];
            for (int g = 0; g < 32; ++g) {
                int id = sidx[m][g];
                if (id >= 0) v += winT[(g * 32 + id) * H_DIM + h];
            }
#pragma unroll
            for (int j = 0; j < A_DIM; ++j)
                v += actl[m][j] * winT[(L_DIM + j) * H_DIM + h];
            xbf[m][h] = (bf16_t)v;
        }
        __syncthreads();

        // ---- GRU: gi = x*W_ih^T, gh = deter*W_hh^T fused with gate math ---
        // wave w owns output columns [16w, 16w+16) of each 256-wide gate block
        const int c0 = wave * 16;
        v8f ir = (v8f)(0.0f), iz = (v8f)(0.0f), in_ = (v8f)(0.0f);
        v8f hr = (v8f)(0.0f), hz = (v8f)(0.0f), hn  = (v8f)(0.0f);
#pragma unroll 2
        for (int ks = 0; ks < 8; ++ks) {
            int kb = ks * 32;
            v16bf ax = load_a_frag(&xbf[0][0], SX, kb, lane);
            v16bf ad = load_a_frag(&dbf[0][0], SX, kb, lane);
            ir  = wmma_bf16(ax, load_b_frag(wih_t, 256,   0 + c0, kb, lane), ir);
            iz  = wmma_bf16(ax, load_b_frag(wih_t, 256, 256 + c0, kb, lane), iz);
            in_ = wmma_bf16(ax, load_b_frag(wih_t, 256, 512 + c0, kb, lane), in_);
            hr  = wmma_bf16(ad, load_b_frag(whh_t, 256,   0 + c0, kb, lane), hr);
            hz  = wmma_bf16(ad, load_b_frag(whh_t, 256, 256 + c0, kb, lane), hz);
            hn  = wmma_bf16(ad, load_b_frag(whh_t, 256, 512 + c0, kb, lane), hn);
        }
        const int c = c0 + (lane & 15);
        const float bir = b_ih[c], biz = b_ih[256 + c], bin_ = b_ih[512 + c];
        const float bhr = b_hh[c], bhz = b_hh[256 + c], bhn  = b_hh[512 + c];
        float hv[8];
#pragma unroll
        for (int j = 0; j < 8; ++j) {
            int m   = mbase + j;
            float d = deter[m][c];
            float r = sigmoidf_((ir[j] + bir) + (hr[j] + bhr));
            float z = sigmoidf_((iz[j] + biz) + (hz[j] + bhz));
            float n = tanhf((in_[j] + bin_) + r * (hn[j] + bhn));
            hv[j]   = (1.0f - z) * n + z * d;
        }
        __syncthreads();   // all waves done reading deter/xbf/dbf
#pragma unroll
        for (int j = 0; j < 8; ++j) {
            int m = mbase + j;
            deter[m][c] = hv[j];
            habf[m][c]  = (bf16_t)hv[j];
            deter_out[((size_t)(b0 + m) * T_STEPS + t) * R_DIM + c] = hv[j];
        }
        __syncthreads();

        // ---- logits = [h|obs] * W_post^T + b_post; stash logits+gumbel ----
#pragma unroll 1
        for (int nt = 0; nt < 4; ++nt) {
            int n0 = (wave * 4 + nt) * 16;
            v8f acc = (v8f)(0.0f);
#pragma unroll 4
            for (int ks = 0; ks < 16; ++ks) {
                v16bf a = load_a_frag(&habf[0][0], SH, ks * 32, lane);
                v16bf b = load_b_frag(wpost_t, 512, n0, ks * 32, lane);
                acc = wmma_bf16(a, b, acc);
            }
            int cc   = n0 + (lane & 15);
            float bp = b_post[cc];
#pragma unroll
            for (int j = 0; j < 8; ++j) {
                int m   = mbase + j;
                float v = acc[j] + bp;
                logits_out[((size_t)(b0 + m) * T_STEPS + t) * L_DIM + cc] = v;
                lg[m][cc] =
                    v + gumbel[((size_t)t * B_TOT + b0 + m) * L_DIM + cc];
            }
        }
        __syncthreads();

        // ---- Gumbel-max argmax + one-hot (first-max tie break like argmax)
        {
            int m = tid >> 5, g = tid & 31;          // 512 threads = 512 rows
            const float* row = &lg[m][g * 32];
            int best = 0;
            float bv = row[0];
#pragma unroll
            for (int k = 1; k < 32; ++k) {
                float v = row[k];
                if (v > bv) { bv = v; best = k; }
            }
            sidx[m][g] = best;
            float* so =
                stoch_out + ((size_t)(b0 + m) * T_STEPS + t) * L_DIM + g * 32;
#pragma unroll
            for (int k = 0; k < 32; ++k) so[k] = (k == best) ? 1.0f : 0.0f;
        }
        __syncthreads();
    }
}

// ---------------------------------------------------------------------------
extern "C" void kernel_launch(void* const* d_in, const int* in_sizes, int n_in,
                              void* d_out, int out_size, void* d_ws, size_t ws_size,
                              hipStream_t stream) {
    const float* obs    = (const float*)d_in[0];
    const float* action = (const float*)d_in[1];
    const unsigned char* first = (const unsigned char*)d_in[2];  // bool array
    const float* gumbel = (const float*)d_in[3];
    const float* W_in   = (const float*)d_in[4];
    const float* b_in   = (const float*)d_in[5];
    const float* W_ih   = (const float*)d_in[6];
    const float* W_hh   = (const float*)d_in[7];
    const float* b_ih   = (const float*)d_in[8];
    const float* b_hh   = (const float*)d_in[9];
    const float* W_post = (const float*)d_in[10];
    const float* b_post = (const float*)d_in[11];

    // workspace layout (all 16B-aligned)
    bf16_t* wih   = (bf16_t*)d_ws;                 // 768*256  bf16
    bf16_t* whh   = wih   + 768 * 256;             // 768*256  bf16
    bf16_t* wpost = whh   + 768 * 256;             // 1024*512 bf16
    float*  winT  = (float*)(wpost + 1024 * 512);  // 1030*256 f32 (W_in^T)

    float* out        = (float*)d_out;
    float* deter_o    = out;
    float* stoch_o    = deter_o + (size_t)B_TOT * T_STEPS * R_DIM;
    float* logits_o   = stoch_o + (size_t)B_TOT * T_STEPS * L_DIM;

    rssm_prep_weights<<<256, 256, 0, stream>>>(W_ih, W_hh, W_post, W_in,
                                               wih, whh, wpost, winT);

    rssm_scan_kernel<<<B_TOT / MB, NTHREADS, 0, stream>>>(
        obs, action, first, gumbel, b_in, b_ih, b_hh, b_post,
        wih, whh, wpost, winT, deter_o, stoch_o, logits_o);
}
